// DecoderAttModule_64630667870869
// MI455X (gfx1250) — matrix operand order repaired and checked by reading, not compiled, not run on hardware
//
#include <hip/hip_runtime.h>
#include <math.h>

// ---------------------------------------------------------------------------
// MI455X (gfx1250) implementation of the Up-Down caption decoder + ARNet.
// - all recurrent weights converted to bf16 once (121 MB -> resident in the
//   192 MB L2 across all 19 timesteps)
// - [W_i | W_h] fused along K so each LSTM is one v_wmma_f32_16x16x32_bf16 GEMM
// - GEMM tiles staged with GLOBAL_LOAD_ASYNC_TO_LDS_B128 (ASYNCcnt), double
//   buffered (unrolled x2 so buffer indices are static), each wave computes a
//   16x64 strip: 1 A fragment feeds 4 independent WMMAs per K-tile
// ---------------------------------------------------------------------------

typedef __bf16 bf16;
typedef __attribute__((ext_vector_type(16))) __bf16 v16bf;
typedef __attribute__((ext_vector_type(8)))  float  v8f;
typedef __attribute__((ext_vector_type(4)))  unsigned u32x4;

#define Bdim 128
#define Tmax 20
#define TdecN 19
#define Vdim 10000
#define Ddim 1024
#define Edim 1024
#define Adim 1024
#define Fdim 2048
#define NG   4096   // 4*D

// GEMM tiling
#define BM 64
#define BN 128
#define BK 32
#define LSTR 40     // padded LDS row stride in bf16 (80B: 16B aligned, conflict-free)

__device__ __forceinline__ float sigf(float x) { return 1.0f / (1.0f + __expf(-x)); }

// Async DMA: global (16B) -> LDS, tracked by ASYNCcnt.
__device__ __forceinline__ void async_ld_b128(const bf16* gptr, bf16* lptr)
{
    const unsigned lds = (unsigned)(unsigned long long)lptr;  // low 32b of flat = LDS offset
    asm volatile("global_load_async_to_lds_b128 %0, %1, off"
                 :: "v"(lds), "v"(gptr)
                 : "memory");
}

// ---------------------------------------------------------------------------
// Core WMMA GEMM:  C[M,N] = A[M,K](bf16) * W[N,K](bf16)^T + bias[N]
// Optional per-row mask: row b zeroed unless declen[b] > t  (vocab projection).
// M must be a multiple of 64; K/32 must be even (all call sites comply).
// Edge N-tiles: W row index clamped to N-1 (garbage accumulates only into
// columns with gn >= N, which the epilogue never stores).
// ---------------------------------------------------------------------------
__global__ __launch_bounds__(256)
void gemm_bf16_wmma(const bf16* __restrict__ A, int lda,
                    const bf16* __restrict__ W, int ldw,
                    const float* __restrict__ bias,
                    float* __restrict__ C, long ldc,
                    int M, int N, int K,
                    const int* __restrict__ declen, int t)
{
    __shared__ __align__(16) bf16 Ash[2][BM * LSTR];
    __shared__ __align__(16) bf16 Wsh[2][BN * LSTR];

    const int tid  = threadIdx.x;
    const int lane = tid & 31;
    const int wid  = tid >> 5;        // 0..7
    const int wr   = wid & 3;         // 0..3 : 16-row band within 64
    const int wcol = wid >> 2;        // 0..1 : 64-col half within 128

    const int m0 = blockIdx.y * BM + wr * 16;
    const int n0 = blockIdx.x * BN + wcol * 64;

    // per-thread DMA slices: 1 A chunk + 2 W chunks of 16B per K-tile
    const int ra = tid >> 2;          // 0..63
    const int cg = tid & 3;           // 0..3 -> 8 bf16

    v8f acc[4];
    #pragma unroll
    for (int s = 0; s < 4; ++s) acc[s] = (v8f){0.f,0.f,0.f,0.f,0.f,0.f,0.f,0.f};

    const bf16* gA = A + (long)(blockIdx.y * BM + ra) * lda + cg * 8;
    const int nr0 = blockIdx.x * BN + ra;
    const int nr1 = nr0 + 64;
    const bf16* gW0 = W + (long)((nr0 < N) ? nr0 : (N - 1)) * ldw + cg * 8;
    const bf16* gW1 = W + (long)((nr1 < N) ? nr1 : (N - 1)) * ldw + cg * 8;

    bf16* lA[2]  = { &Ash[0][ra * LSTR + cg * 8],        &Ash[1][ra * LSTR + cg * 8] };
    bf16* lW0[2] = { &Wsh[0][ra * LSTR + cg * 8],        &Wsh[1][ra * LSTR + cg * 8] };
    bf16* lW1[2] = { &Wsh[0][(64 + ra) * LSTR + cg * 8], &Wsh[1][(64 + ra) * LSTR + cg * 8] };

    auto issue = [&](int buf, int k) {
        async_ld_b128(gA  + k, lA[buf]);
        async_ld_b128(gW0 + k, lW0[buf]);
        async_ld_b128(gW1 + k, lW1[buf]);
    };

    auto compute = [&](int buf) {
        // A fragment (ISA 16-bit A 16x32 layout):
        // lane<16: row=lane,    K in {0..7, 16..23}
        // lane>=16: row=lane-16, K in {8..15, 24..31}, 2 values per VGPR
        union { v16bf v; unsigned u[8]; } af;
        const bf16* As = Ash[buf];
        const bf16* Ws = Wsh[buf];
        const int arow = wr * 16 + (lane & 15);
        const int kb   = (lane >> 4) * 8;
        #pragma unroll
        for (int j = 0; j < 8; ++j) {
            const int kk = ((j & 4) ? 16 : 0) + kb + 2 * (j & 3);
            af.u[j] = *(const unsigned*)(As + arow * LSTR + kk);
        }
        // B fragments: lane<16 holds col=lane K in {0..7, 16..23}; lane>=16 K in {8..15, 24..31}
        const int kb2 = (lane >> 4) * 8;
        #pragma unroll
        for (int s = 0; s < 4; ++s) {
            union { v16bf v; u32x4 q[2]; } bfr;
            const bf16* p = Ws + (wcol * 64 + s * 16 + (lane & 15)) * LSTR + kb2;
            bfr.q[0] = *(const u32x4*)(p);
            bfr.q[1] = *(const u32x4*)(p + 16);
            acc[s] = __builtin_amdgcn_wmma_f32_16x16x32_bf16(false, af.v, false, bfr.v,
                                                             (short)0, acc[s], false, false);
        }
    };

    // prologue: tile 0 in flight
    issue(0, 0);

    for (int k0 = 0; k0 < K; k0 += 2 * BK) {
        // phase 0: prefetch tile k0+BK into buf1 (always exists: K/BK even),
        // wait for tile k0, compute from buf0
        issue(1, k0 + BK);
        asm volatile("s_wait_asynccnt 0x3" ::: "memory");
        __syncthreads();
        compute(0);
        __syncthreads();

        // phase 1: prefetch tile k0+2BK into buf0 if any, compute from buf1
        if (k0 + 2 * BK < K) {
            issue(0, k0 + 2 * BK);
            asm volatile("s_wait_asynccnt 0x3" ::: "memory");
        } else {
            asm volatile("s_wait_asynccnt 0x0" ::: "memory");
        }
        __syncthreads();
        compute(1);
        __syncthreads();
    }

    // epilogue: C layout VGPR r -> M = r + 8*(lane>>4), N = lane&15
    const int mrow = m0 + 8 * (lane >> 4);
    const int gnb  = n0 + (lane & 15);
    #pragma unroll
    for (int s = 0; s < 4; ++s) {
        const int gn = gnb + s * 16;
        if (gn < N) {
            const float bv = bias ? bias[gn] : 0.0f;
            #pragma unroll
            for (int r = 0; r < 8; ++r) {
                const int gm = mrow + r;
                float scale = 1.0f;
                if (declen) scale = (declen[gm] > t) ? 1.0f : 0.0f;
                C[(long)gm * ldc + gn] = scale * (acc[s][r] + bv);
            }
        }
    }
}

// ---------------------------------------------------------------------------
// Setup kernels
// ---------------------------------------------------------------------------

// Stable descending sort (rank-based, B=128 fits one block).
__global__ void sort_setup(const int* __restrict__ sizes,
                           const int* __restrict__ sequences,
                           int* __restrict__ order,
                           int* __restrict__ declen,
                           int* __restrict__ seqs_s)
{
    __shared__ int ssz[Bdim];
    __shared__ int sord[Bdim];
    const int b = threadIdx.x;
    ssz[b] = sizes[b];
    __syncthreads();
    const int key = ssz[b];
    int rank = 0;
    for (int j = 0; j < Bdim; ++j) {
        const int kj = ssz[j];
        if (kj > key || (kj == key && j < b)) rank++;
    }
    sord[rank] = b;
    __syncthreads();
    const int o = sord[b];
    order[b]  = o;
    declen[b] = ssz[o] - 1;
    for (int t2 = 0; t2 < Tmax; ++t2)
        seqs_s[b * Tmax + t2] = sequences[o * Tmax + t2];
}

__global__ void zero_u32(unsigned* __restrict__ p, long n)
{
    long i = (long)blockIdx.x * 256 + threadIdx.x;
    if (i < n) p[i] = 0u;
}

__global__ void f32_to_bf16_k(const float* __restrict__ src, bf16* __restrict__ dst, long n)
{
    long i = (long)blockIdx.x * 256 + threadIdx.x;
    if (i < n) dst[i] = (bf16)src[i];
}

// dst[n, 0:Ka] = A[n,:], dst[n, Ka:Ka+Kb] = Bsrc[n,:]   (bf16)
__global__ void concat2_bf16_k(const float* __restrict__ A, int Ka,
                               const float* __restrict__ Bsrc, int Kb,
                               bf16* __restrict__ dst, long n)
{
    long i = (long)blockIdx.x * 256 + threadIdx.x;
    if (i >= n) return;
    const int Kt = Ka + Kb;
    const long r = i / Kt;
    const int  k = (int)(i - r * Kt);
    const float v = (k < Ka) ? A[r * Ka + k] : Bsrc[r * Kb + (k - Ka)];
    dst[i] = (bf16)v;
}

__global__ void add_bias_k(const float* __restrict__ a, const float* __restrict__ b,
                           float* __restrict__ o, int n)
{
    int i = blockIdx.x * 256 + threadIdx.x;
    if (i < n) o[i] = a[i] + b[i];
}

// featsAvg -> bf16 slice of xtd at column offset D
__global__ void feats_avg_k(const float* __restrict__ feats, const int* __restrict__ order,
                            bf16* __restrict__ xtd)
{
    const int idx = blockIdx.x * 256 + threadIdx.x;   // B*F
    const int b = idx / Fdim, f = idx % Fdim;
    const int o = order[b];
    const float* fb = feats + (long)o * 36 * Fdim + f;
    float s = 0.f;
    #pragma unroll 4
    for (int r = 0; r < 36; ++r) s += fb[(long)r * Fdim];
    xtd[(long)b * 5120 + Ddim + f] = (bf16)(s * (1.0f / 36.0f));
}

__global__ void feats_sort_k(const float* __restrict__ feats, const int* __restrict__ order,
                             bf16* __restrict__ feats_s)
{
    long idx = (long)blockIdx.x * 256 + threadIdx.x;  // B*36*F
    if (idx >= (long)Bdim * 36 * Fdim) return;
    const long per = (long)36 * Fdim;
    const int b = (int)(idx / per);
    const long rem = idx - (long)b * per;
    feats_s[idx] = (bf16)feats[(long)order[b] * per + rem];
}

// ---------------------------------------------------------------------------
// Per-step kernels
// ---------------------------------------------------------------------------

__global__ void emb_gather_k(const float* __restrict__ emb, const int* __restrict__ seqs_s,
                             int t, bf16* __restrict__ xtd)
{
    const int idx = blockIdx.x * 256 + threadIdx.x;   // B*E
    const int b = idx >> 10, e = idx & 1023;
    const int tok = seqs_s[b * Tmax + t];
    xtd[(long)b * 5120 + (Ddim + Fdim) + e] = (bf16)emb[(long)tok * Edim + e];
}

// gates[B,4096] (i,f,g,o) -> masked LSTM state update; mirrors new h as bf16
// into up to three x-buffer slices for the downstream GEMMs / next step.
__global__ __launch_bounds__(256)
void lstm_update_k(const float* __restrict__ gates,
                   float* __restrict__ h, float* __restrict__ c,
                   float* __restrict__ prev_save,
                   const int* __restrict__ declen, int t,
                   bf16* d0, long s0, bf16* d1, long s1, bf16* d2, long s2)
{
    const int idx = blockIdx.x * 256 + threadIdx.x;   // B*D
    const int b = idx >> 10;
    const int d = idx & 1023;
    const float* g = gates + (long)b * NG;
    const float gi = g[d], gf = g[Ddim + d], gg = g[2 * Ddim + d], go = g[3 * Ddim + d];
    const float cold = c[idx], hold = h[idx];
    const float cn = sigf(gf) * cold + sigf(gi) * tanhf(gg);
    const float hn = sigf(go) * tanhf(cn);
    const float m  = (declen[b] > t) ? 1.f : 0.f;
    const float hv = m * hn + (1.f - m) * hold;
    const float cv = m * cn + (1.f - m) * cold;
    if (prev_save) prev_save[idx] = hold;
    h[idx] = hv;
    c[idx] = cv;
    const bf16 hb = (bf16)hv;
    if (d0) d0[(long)b * s0 + d] = hb;
    if (d1) d1[(long)b * s1 + d] = hb;
    if (d2) d2[(long)b * s2 + d] = hb;
}

// Visual attention: e = relu(h1a + img_att) @ aw_w + aw_b; softmax over 36;
// awe = alpha @ feats; write awe as bf16 into xlg[:,0:2048]. One block per b.
__global__ __launch_bounds__(256)
void attention_k(const float* __restrict__ h1a,
                 const float* __restrict__ img_att,
                 const bf16*  __restrict__ feats_s,
                 const float* __restrict__ aw_w,
                 const float* __restrict__ aw_b,
                 bf16* __restrict__ awe_dst, long awe_stride)
{
    __shared__ float red[256];
    __shared__ float ebuf[36];
    __shared__ float alpha[36];
    const int b = blockIdx.x;
    const int tid = threadIdx.x;
    const float* h1ab = h1a + (long)b * Adim;
    const float* iab  = img_att + (long)b * 36 * Adim;

    for (int r = 0; r < 36; ++r) {
        float p = 0.f;
        const float* ia = iab + (long)r * Adim;
        for (int a = tid; a < Adim; a += 256) {
            float v = h1ab[a] + ia[a];
            v = v > 0.f ? v : 0.f;
            p += v * aw_w[a];
        }
        red[tid] = p;
        __syncthreads();
        for (int s = 128; s > 0; s >>= 1) {
            if (tid < s) red[tid] += red[tid + s];
            __syncthreads();
        }
        if (tid == 0) ebuf[r] = red[0] + aw_b[0];
        __syncthreads();
    }
    if (tid == 0) {
        float mx = ebuf[0];
        for (int r = 1; r < 36; ++r) mx = fmaxf(mx, ebuf[r]);
        float s = 0.f;
        for (int r = 0; r < 36; ++r) { const float e2 = __expf(ebuf[r] - mx); alpha[r] = e2; s += e2; }
        const float inv = 1.f / s;
        for (int r = 0; r < 36; ++r) alpha[r] *= inv;
    }
    __syncthreads();
    const bf16* fb = feats_s + (long)b * 36 * Fdim;
    bf16* outb = awe_dst + (long)b * awe_stride;
    for (int f = tid; f < Fdim; f += 256) {
        float s = 0.f;
        #pragma unroll 6
        for (int r = 0; r < 36; ++r) s += alpha[r] * (float)fb[(long)r * Fdim + f];
        outb[f] = (bf16)s;
    }
}

// loss += sum_b m_b * ||arl(arh) - prev_h1||^2 / bs * 0.005   (single block: deterministic)
__global__ __launch_bounds__(256)
void arnet_loss_k(const float* __restrict__ arl_out, const float* __restrict__ prev_h1,
                  const int* __restrict__ declen, int t, float* __restrict__ loss_acc)
{
    __shared__ float red[256];
    const int tid = threadIdx.x;
    float total = 0.f;
    for (int idx = tid; idx < Bdim * Ddim; idx += 256) {
        const int b = idx >> 10;
        const float m = (declen[b] > t) ? 1.f : 0.f;
        const float dlt = arl_out[idx] - prev_h1[idx];
        total += m * dlt * dlt;
    }
    red[tid] = total;
    __syncthreads();
    for (int s = 128; s > 0; s >>= 1) {
        if (tid < s) red[tid] += red[tid + s];
        __syncthreads();
    }
    if (tid == 0) {
        float cnt = 0.f;
        for (int b = 0; b < Bdim; ++b) cnt += (declen[b] > t) ? 1.f : 0.f;
        *loss_acc += red[0] / cnt * 0.005f;
    }
}

// Tail outputs: seqs (sorted) as float, order as float, loss scalar.
__global__ void finalize_k(const int* __restrict__ seqs_s, const int* __restrict__ order,
                           const float* __restrict__ loss_acc, float* __restrict__ out)
{
    const int tid = blockIdx.x * 256 + threadIdx.x;
    const long base = (long)Bdim * TdecN * Vdim;
    if (tid < Bdim * Tmax) out[base + tid] = (float)seqs_s[tid];
    if (tid < Bdim)        out[base + Bdim * Tmax + tid] = (float)order[tid];
    if (tid == 0)          out[base + Bdim * Tmax + Bdim] = *loss_acc;
}

// ---------------------------------------------------------------------------
// Host orchestration
// ---------------------------------------------------------------------------
extern "C" void kernel_launch(void* const* d_in, const int* in_sizes, int n_in,
                              void* d_out, int out_size, void* d_ws, size_t ws_size,
                              hipStream_t stream)
{
    (void)in_sizes; (void)n_in; (void)out_size; (void)ws_size;

    const float* feats      = (const float*)d_in[0];
    const int*   sequences  = (const int*)  d_in[1];
    const int*   sizes      = (const int*)  d_in[2];
    const float* emb        = (const float*)d_in[3];
    const float* td_wi      = (const float*)d_in[4];
    const float* td_wh      = (const float*)d_in[5];
    const float* td_bi      = (const float*)d_in[6];
    const float* td_bh      = (const float*)d_in[7];
    const float* lg_wi      = (const float*)d_in[8];
    const float* lg_wh      = (const float*)d_in[9];
    const float* lg_bi      = (const float*)d_in[10];
    const float* lg_bh      = (const float*)d_in[11];
    const float* ar_wi      = (const float*)d_in[12];
    const float* ar_wh      = (const float*)d_in[13];
    const float* ar_bi      = (const float*)d_in[14];
    const float* ar_bh      = (const float*)d_in[15];
    const float* arl_w      = (const float*)d_in[16];
    const float* arl_b      = (const float*)d_in[17];
    const float* out_w      = (const float*)d_in[18];
    const float* out_b      = (const float*)d_in[19];
    const float* af_w       = (const float*)d_in[20];
    const float* af_b       = (const float*)d_in[21];
    const float* ad_w       = (const float*)d_in[22];
    const float* ad_b       = (const float*)d_in[23];
    const float* aw_w       = (const float*)d_in[24];
    const float* aw_b       = (const float*)d_in[25];

    float* out_f = (float*)d_out;

    // ---- workspace layout ----
    char* ws = (char*)d_ws;
    size_t off = 0;
    auto alloc = [&](size_t bytes) -> char* {
        char* p = ws + off;
        off = (off + bytes + 255) & ~(size_t)255;
        return p;
    };
    int*   order   = (int*)  alloc(Bdim * 4);
    int*   declen  = (int*)  alloc(Bdim * 4);
    int*   seqs_s  = (int*)  alloc(Bdim * Tmax * 4);
    float* loss    = (float*)alloc(4);
    bf16*  xtd     = (bf16*) alloc((size_t)Bdim * 5120 * 2);  // [h2|featsAvg|emb|h1]
    bf16*  xlg     = (bf16*) alloc((size_t)Bdim * 4096 * 2);  // [awe|h1|h2_old]
    bf16*  xar     = (bf16*) alloc((size_t)Bdim * 2048 * 2);  // [h1|arh_old]
    float* gates   = (float*)alloc((size_t)Bdim * NG * 4);
    float* states  = (float*)alloc((size_t)7 * Bdim * Ddim * 4);
    float* h1 = states + 0 * Bdim * Ddim;
    float* c1 = states + 1 * Bdim * Ddim;
    float* h2 = states + 2 * Bdim * Ddim;
    float* c2 = states + 3 * Bdim * Ddim;
    float* arh = states + 4 * Bdim * Ddim;
    float* arc = states + 5 * Bdim * Ddim;
    float* prevh1 = states + 6 * Bdim * Ddim;
    float* h1a     = (float*)alloc((size_t)Bdim * Adim * 4);
    float* arl_out = (float*)alloc((size_t)Bdim * Ddim * 4);
    float* btd     = (float*)alloc(NG * 4);
    float* blg     = (float*)alloc(NG * 4);
    float* bar     = (float*)alloc(NG * 4);
    bf16*  feats_s = (bf16*) alloc((size_t)Bdim * 36 * Fdim * 2);
    float* img_att = (float*)alloc((size_t)Bdim * 36 * Adim * 4);
    bf16*  Wtd     = (bf16*) alloc((size_t)NG * 5120 * 2);
    bf16*  Wlg     = (bf16*) alloc((size_t)NG * 4096 * 2);
    bf16*  War     = (bf16*) alloc((size_t)NG * 2048 * 2);
    bf16*  Warl    = (bf16*) alloc((size_t)Ddim * Ddim * 2);
    bf16*  Wout    = (bf16*) alloc((size_t)Vdim * Ddim * 2);
    bf16*  Wad     = (bf16*) alloc((size_t)Adim * Ddim * 2);
    bf16*  Waf     = (bf16*) alloc((size_t)Adim * Fdim * 2);

    auto gblk = [](long n) -> unsigned { return (unsigned)((n + 255) / 256); };

    // ---- setup ----
    sort_setup<<<1, Bdim, 0, stream>>>(sizes, sequences, order, declen, seqs_s);

    zero_u32<<<gblk(7L * Bdim * Ddim), 256, 0, stream>>>((unsigned*)states, 7L * Bdim * Ddim);
    zero_u32<<<1, 256, 0, stream>>>((unsigned*)loss, 1);
    zero_u32<<<gblk((long)Bdim * 5120 / 2), 256, 0, stream>>>((unsigned*)xtd, (long)Bdim * 5120 / 2);
    zero_u32<<<gblk((long)Bdim * 4096 / 2), 256, 0, stream>>>((unsigned*)xlg, (long)Bdim * 4096 / 2);
    zero_u32<<<gblk((long)Bdim * 2048 / 2), 256, 0, stream>>>((unsigned*)xar, (long)Bdim * 2048 / 2);

    feats_avg_k<<<gblk((long)Bdim * Fdim), 256, 0, stream>>>(feats, order, xtd);
    feats_sort_k<<<gblk((long)Bdim * 36 * Fdim), 256, 0, stream>>>(feats, order, feats_s);

    concat2_bf16_k<<<gblk((long)NG * 5120), 256, 0, stream>>>(td_wi, 4096, td_wh, 1024, Wtd, (long)NG * 5120);
    concat2_bf16_k<<<gblk((long)NG * 4096), 256, 0, stream>>>(lg_wi, 3072, lg_wh, 1024, Wlg, (long)NG * 4096);
    concat2_bf16_k<<<gblk((long)NG * 2048), 256, 0, stream>>>(ar_wi, 1024, ar_wh, 1024, War, (long)NG * 2048);
    f32_to_bf16_k<<<gblk((long)Ddim * Ddim), 256, 0, stream>>>(arl_w, Warl, (long)Ddim * Ddim);
    f32_to_bf16_k<<<gblk((long)Vdim * Ddim), 256, 0, stream>>>(out_w, Wout, (long)Vdim * Ddim);
    f32_to_bf16_k<<<gblk((long)Adim * Ddim), 256, 0, stream>>>(ad_w, Wad, (long)Adim * Ddim);
    f32_to_bf16_k<<<gblk((long)Adim * Fdim), 256, 0, stream>>>(af_w, Waf, (long)Adim * Fdim);
    add_bias_k<<<gblk(NG), 256, 0, stream>>>(td_bi, td_bh, btd, NG);
    add_bias_k<<<gblk(NG), 256, 0, stream>>>(lg_bi, lg_bh, blg, NG);
    add_bias_k<<<gblk(NG), 256, 0, stream>>>(ar_bi, ar_bh, bar, NG);

    // img_att[B*36, A] = feats_s[B*36, F] @ af_w^T + af_b
    gemm_bf16_wmma<<<dim3(Adim / BN, (Bdim * 36) / BM), 256, 0, stream>>>(
        feats_s, Fdim, Waf, Fdim, af_b, img_att, Adim, Bdim * 36, Adim, Fdim, nullptr, 0);

    // ---- recurrence ----
    for (int t = 0; t < TdecN; ++t) {
        // top-down LSTM: xtd = [h2_old | featsAvg | emb_t | h1_old]
        emb_gather_k<<<gblk((long)Bdim * Edim), 256, 0, stream>>>(emb, seqs_s, t, xtd);
        gemm_bf16_wmma<<<dim3(NG / BN, Bdim / BM), 256, 0, stream>>>(
            xtd, 5120, Wtd, 5120, btd, gates, NG, Bdim, NG, 5120, nullptr, 0);
        lstm_update_k<<<gblk((long)Bdim * Ddim), 256, 0, stream>>>(
            gates, h1, c1, prevh1, declen, t,
            xtd + 4096, 5120, xlg + 2048, 4096, xar, 2048);

        // attention
        gemm_bf16_wmma<<<dim3(Adim / BN, Bdim / BM), 256, 0, stream>>>(
            xar, 2048, Wad, Ddim, ad_b, h1a, Adim, Bdim, Adim, Ddim, nullptr, 0);
        attention_k<<<Bdim, 256, 0, stream>>>(h1a, img_att, feats_s, aw_w, aw_b, xlg, 4096);

        // language LSTM: xlg = [awe | h1_new | h2_old]
        gemm_bf16_wmma<<<dim3(NG / BN, Bdim / BM), 256, 0, stream>>>(
            xlg, 4096, Wlg, 4096, blg, gates, NG, Bdim, NG, 4096, nullptr, 0);
        lstm_update_k<<<gblk((long)Bdim * Ddim), 256, 0, stream>>>(
            gates, h2, c2, nullptr, declen, t,
            xtd, 5120, xlg + 3072, 4096, nullptr, 0);

        // ARNet LSTM: xar = [h1_new | arh_old]
        gemm_bf16_wmma<<<dim3(NG / BN, Bdim / BM), 256, 0, stream>>>(
            xar, 2048, War, 2048, bar, gates, NG, Bdim, NG, 2048, nullptr, 0);
        lstm_update_k<<<gblk((long)Bdim * Ddim), 256, 0, stream>>>(
            gates, arh, arc, nullptr, declen, t,
            xar + 1024, 2048, nullptr, 0, nullptr, 0);

        // reconstruction loss
        gemm_bf16_wmma<<<dim3(Ddim / BN, Bdim / BM), 256, 0, stream>>>(
            xar + 1024, 2048, Warl, Ddim, arl_b, arl_out, Ddim, Bdim, Ddim, Ddim, nullptr, 0);
        if (t > 0)
            arnet_loss_k<<<1, 256, 0, stream>>>(arl_out, prevh1, declen, t, loss);

        // vocab projection (masked), strided into preds[B, Tdec, V]
        gemm_bf16_wmma<<<dim3((Vdim + BN - 1) / BN, Bdim / BM), 256, 0, stream>>>(
            xtd, 5120, Wout, Ddim, out_b, out_f + (long)t * Vdim, (long)TdecN * Vdim,
            Bdim, Vdim, Ddim, declen, t);
    }

    finalize_k<<<gblk(Bdim * Tmax), 256, 0, stream>>>(seqs_s, order, loss, out_f);
}
